// NVFP4Linear_12360915878779
// MI455X (gfx1250) — matrix operand order repaired and checked by reading, not compiled, not run on hardware
//
#include <hip/hip_runtime.h>

// ---------------------------------------------------------------------------
// NVFP4 linear: out[128,8192] = QDQ(A)[128,8192] * DQ(W)[8192,8192]^T
// Weight-bandwidth bound (~157MB @ 23.3TB/s ~ 6.7us floor) -> read weight once,
// use v_wmma_f32_16x16x32_f16 (fp16 dequant of fp4*e4m3 is exact in fp16).
// Round 4: LUT-free e2m1 decode -- nibble bits dropped straight into the fp16
// exponent/mantissa field give e2m1_value * 2^-14 exactly (denorms preserved),
// with the 2^14 folded into the per-group scale. ~5 VALU ops per weight byte.
// ---------------------------------------------------------------------------

typedef __attribute__((ext_vector_type(16))) _Float16 v16h;
typedef __attribute__((ext_vector_type(8)))  _Float16 v8h;
typedef __attribute__((ext_vector_type(8)))  float    v8f;
typedef __attribute__((ext_vector_type(8)))  int      v8i;

#define M_DIM   128
#define N_DIM   8192
#define K_DIM   8192
#define W_INTS  (K_DIM / 2)    // int32 elements per weight row (one byte each)
#define NGROUPS (K_DIM / 16)   // 512 scale groups per row

// ---- round float (>0) to nearest fp8 e4m3 value (RNE), emulated in bits ----
__device__ __forceinline__ float e4m3_rne(float v) {
    if (!(v > 0.0f)) return v;              // amax>=0; 0 stays 0
    v = fminf(v, 448.0f);
    unsigned u = __float_as_uint(v);
    int e = (int)((u >> 23) & 0xFF) - 127;
    if (e < -6) {                           // e4m3 subnormal: quantum 2^-9
        return rintf(v * 512.0f) * (1.0f / 512.0f);
    }
    unsigned lsb = (u >> 20) & 1u;          // keep 3 mantissa bits
    u += ((1u << 19) - 1u + lsb);           // round-to-nearest-even
    u &= ~((1u << 20) - 1u);
    return fminf(__uint_as_float(u), 448.0f);
}

// ---- round to nearest e2m1 grid value (ties match searchsorted 'left') ----
__device__ __forceinline__ float e2m1_round(float x) {
    float a = fabsf(x);
    float v = (a <= 0.25f) ? 0.0f :
              (a <= 0.75f) ? 0.5f :
              (a <= 1.25f) ? 1.0f :
              (a <= 1.75f) ? 1.5f :
              (a <= 2.50f) ? 2.0f :
              (a <= 3.50f) ? 3.0f :
              (a <= 5.00f) ? 4.0f : 6.0f;
    return (x < 0.0f) ? -v : v;
}

// ---- LUT-free e2m1 byte -> packed 2xfp16 (each = value * 2^-14) ----
// nibble [s|e1e0|m]: (nib&7)<<9 places e in fp16 exp[10:9... i.e. exp LSBs]
// and m in mant MSB => (1+m/2)*2^(e-15) = value*2^-14 (e>0), and the e==0
// case lands on the fp16 denormal m*2^-15 = (m*0.5)*2^-14. Sign: bit3 -> 15.
__device__ __forceinline__ unsigned e2m1_pair_bits(unsigned byte) {
    const unsigned t = byte | (byte << 12);   // nib0 at [3:0], nib1 at [19:16]
    return ((t & 0x00070007u) << 9) | ((t & 0x00080008u) << 12);
}

// ---- async global->LDS copy of 32B (ASYNCcnt-tracked) ----
__device__ __forceinline__ void async_copy32(unsigned ldsOff, const _Float16* gp) {
    asm volatile(
        "global_load_async_to_lds_b128 %0, %1, off\n\t"
        "global_load_async_to_lds_b128 %0, %1, off offset:16"
        :: "v"(ldsOff), "v"(gp) : "memory");
}
__device__ __forceinline__ void wait_async0() {
    asm volatile("s_wait_asynccnt 0x0" ::: "memory");
}

// ===========================================================================
// Kernel 1: per-16-group dynamic quantize/dequantize of activations -> fp16 A
// one thread per group; 65536 groups total
// ===========================================================================
__global__ void nvfp4_act_qdq_kernel(const _Float16* __restrict__ x,
                                     const float* __restrict__ input_scale,
                                     _Float16* __restrict__ aq) {
    const int gid = blockIdx.x * blockDim.x + threadIdx.x;   // [0, 65536)
    const v16h xv = *(const v16h*)(x + (size_t)gid * 16);
    const float isc = input_scale[0];

    float xf[16];
    float amax = 0.0f;
#pragma unroll
    for (int i = 0; i < 16; ++i) {
        xf[i] = (float)xv[i];
        amax  = fmaxf(amax, fabsf(xf[i]));
    }
    float s = e4m3_rne(amax / (6.0f * isc));
    s = (s > 0.0f) ? s : 1.0f;
    const float tot = s * isc;

    v16h o;
#pragma unroll
    for (int i = 0; i < 16; ++i)
        o[i] = (_Float16)(e2m1_round(xf[i] / tot) * tot);
    *(v16h*)(aq + (size_t)gid * 16) = o;
}

// ===========================================================================
// Kernel 2: GEMM. 128 blocks x 256 threads (8 waves).
// Block tile: 128(M) x 64(N). Waves: 4 N-strips x 2 M-halves, 4 acc tiles each.
// K loop step 32: A double-buffered via async LDS copy, W decoded in-register.
// ===========================================================================
__global__ void __launch_bounds__(256)
nvfp4_gemm_kernel(const _Float16* __restrict__ aq,      // [128, 8192] fp16
                  const int*      __restrict__ w,       // [8192, 4096] one byte/int
                  const float*    __restrict__ wscale,  // [8192, 512]
                  const float*    __restrict__ ws2p,    // [1]
                  float*          __restrict__ out) {   // [128, 8192] fp32
    __shared__ __align__(32) _Float16 As[2][M_DIM * 32];   // 2 x 8 KB A stage

    const int tid   = threadIdx.x;
    const int wave  = tid >> 5;
    const int lane  = tid & 31;
    const int nq    = wave & 3;        // N strip within block
    const int mhalf = wave >> 2;       // M half (tiles 0..3 or 4..7)
    const int nbase = blockIdx.x * 64;
    const int nglob = nbase + nq * 16 + (lane & 15);    // B column for this lane

    const float ws2 = ws2p[0];
    const size_t wrow = (size_t)nglob * W_INTS;
    const size_t srow = (size_t)nglob * NGROUPS;

    v8f acc[4] = {};   // 4 M-tiles of 16x16 fp32

    const int arow = lane & 15;          // A fragment row within tile
    const int ak   = (lane >> 4) * 16;   // A fragment K sub-chunk (0 or 16)

    // per-thread A staging addresses (32B per thread per k-step)
    const int srow_id = tid >> 1;
    const int scol0   = (tid & 1) * 16;
    const unsigned ldsOff0 = (unsigned)(uintptr_t)&As[0][srow_id * 32 + scol0];
    const unsigned ldsOff1 = (unsigned)(uintptr_t)&As[1][srow_id * 32 + scol0];
    const _Float16* gbase  = aq + (size_t)srow_id * K_DIM + scol0;

    // prologue: fill buffer 0 with chunk kk=0
    async_copy32(ldsOff0, gbase);
    wait_async0();
    __syncthreads();

    for (int kk = 0; kk < K_DIM; kk += 32) {
        const int cur = (kk >> 5) & 1;

        // -- kick off async copy of the NEXT A chunk + prefetch next weights --
        if (kk + 32 < K_DIM) {
            async_copy32(cur ? ldsOff0 : ldsOff1, gbase + kk + 32);
            __builtin_prefetch(w + wrow + ((kk + 32 + ((lane >> 4) << 4)) >> 1), 0, 1);
        }

        // -- B fragment: lane holds column nglob, K = kb..kb+15 (one group) --
        const int kb = kk + ((lane >> 4) << 4);
        const v8i codes = *(const v8i*)(w + wrow + (kb >> 1));
        // per-group scale with the 2^14 decode factor folded in
        const _Float16 sch = (_Float16)(wscale[srow + (kb >> 4)] * ws2 * 16384.0f);

        union { unsigned u[8]; v16h h; } bb;
#pragma unroll
        for (int j = 0; j < 8; ++j)
            bb.u[j] = e2m1_pair_bits((unsigned)codes[j] & 0xFFu);

        v16h bfrag;
#pragma unroll
        for (int i = 0; i < 16; ++i)
            bfrag[i] = bb.h[i] * sch;

        // -- load all 4 A fragments into distinct regs (one ds clause) --
        v8h lo[4], hi[4];
#pragma unroll
        for (int t = 0; t < 4; ++t) {
            const int r = (mhalf * 4 + t) * 16 + arow;
            lo[t] = *(const v8h*)&As[cur][r * 32 + ak];
            hi[t] = *(const v8h*)&As[cur][r * 32 + ak + 8];
        }

        // -- 4 back-to-back WMMAs --
#pragma unroll
        for (int t = 0; t < 4; ++t) {
            const v16h afrag = __builtin_shufflevector(
                lo[t], hi[t], 0, 1, 2, 3, 4, 5, 6, 7, 8, 9, 10, 11, 12, 13, 14, 15);
            acc[t] = __builtin_amdgcn_wmma_f32_16x16x32_f16(
                false, afrag, false, bfrag, (short)0, acc[t], false, false);
        }

        // -- close the iteration: next-chunk writes landed, all reads done --
        wait_async0();
        __syncthreads();
    }

    // -- write out: C layout: VGPR r -> M = tile*16 + r + 8*(lane/16), N = lane%16
#pragma unroll
    for (int t = 0; t < 4; ++t) {
#pragma unroll
        for (int r = 0; r < 8; ++r) {
            const int m = (mhalf * 4 + t) * 16 + r + ((lane >> 4) << 3);
            out[(size_t)m * N_DIM + nglob] = acc[t][r];
        }
    }
}

// ===========================================================================
extern "C" void kernel_launch(void* const* d_in, const int* in_sizes, int n_in,
                              void* d_out, int out_size, void* d_ws, size_t ws_size,
                              hipStream_t stream) {
    const _Float16* hidden      = (const _Float16*)d_in[0];  // [8,16,8192] fp16
    const int*      weight      = (const int*)d_in[1];       // [8192,4096] int32
    const float*    wscale      = (const float*)d_in[2];     // [8192,512]
    const float*    wscale2     = (const float*)d_in[3];     // [1]
    const float*    input_scale = (const float*)d_in[4];     // [1]
    float*          out         = (float*)d_out;             // [128,8192]

    _Float16* aq = (_Float16*)d_ws;                          // 2 MB staged A

    // 1) activation QDQ: 65536 groups of 16
    nvfp4_act_qdq_kernel<<<(M_DIM * NGROUPS) / 256, 256, 0, stream>>>(
        hidden, input_scale, aq);

    // 2) GEMM: 128 blocks over N (64 cols each), weight streamed exactly once
    nvfp4_gemm_kernel<<<N_DIM / 64, 256, 0, stream>>>(
        aq, weight, wscale, wscale2, out);
}
